// EditOuterAttention_88476326297687
// MI455X (gfx1250) — matrix operand order repaired and checked by reading, not compiled, not run on hardware
//
#include <hip/hip_runtime.h>

// Problem constants (from reference)
#define BS  2
#define LX  2048
#define LY  2048
#define DD  1024     // model dim
#define HH  16       // heads
#define DKK 64       // head dim
#define HD  1024     // H*DK

typedef __attribute__((ext_vector_type(16))) __bf16 v16bf;
typedef __attribute__((ext_vector_type(8)))  __bf16 v8bf;
typedef __attribute__((ext_vector_type(8)))  float  v8f;

union BFrag { v16bf v; v8bf h[2]; };

__device__ __forceinline__ __bf16 f2bf(float f) {
  unsigned u = __builtin_bit_cast(unsigned, f);
  unsigned r = u + 0x7FFFu + ((u >> 16) & 1u);   // round-to-nearest-even
  unsigned short s = (unsigned short)(r >> 16);
  return __builtin_bit_cast(__bf16, s);
}
__device__ __forceinline__ unsigned bfbits(__bf16 b) {
  return (unsigned)__builtin_bit_cast(unsigned short, b);
}

__device__ __forceinline__ v8f wmma_bf16(const BFrag& a, const BFrag& b, v8f c) {
  // (neg_a, A, neg_b, B, c_mod, C, reuse_a, reuse_b)
  return __builtin_amdgcn_wmma_f32_16x16x32_bf16(false, a.v, false, b.v, (short)0, c, false, false);
}

// Async copy 16B from global to LDS (per lane), tracked by ASYNCcnt.
__device__ __forceinline__ void async_g2l_b128(unsigned lds_off, const void* gaddr) {
  asm volatile("global_load_async_to_lds_b128 %0, %1, off"
               :: "v"(lds_off), "v"((unsigned long long)(uintptr_t)gaddr)
               : "memory");
}
__device__ __forceinline__ void wait_async0() {
  asm volatile("s_wait_asynccnt 0x0" ::: "memory");
}
__device__ __forceinline__ unsigned lds_off_of(const void* p) {
  // addrspace(3) offset == low 32 bits of the generic LDS address
  return (unsigned)(uintptr_t)p;
}

// ---------------------------------------------------------------------------
// Tiled GEMM:  C[b,M,N] = A[b,M,K] @ W[K,N] + bias[N]
// A is f32 (converted to bf16) or bf16 (staged via async-to-LDS DMA);
// C is bf16 (to workspace) or f32 (final).
// Block = 256 threads (8 waves, 2x4 wave grid), tile 128x128, K-step 32.
// ---------------------------------------------------------------------------
template<bool A_BF16, bool OUT_F32>
__global__ __launch_bounds__(256) void gemm_kernel(
    const void* __restrict__ Av, const float* __restrict__ W,
    const float* __restrict__ bias, void* __restrict__ Cv,
    int M, int N, int K)
{
  __shared__ __bf16 sA[128][40];   // [m][k]   (pad 32->40 halfs, 80B pitch)
  __shared__ __bf16 sB[128][40];   // [n][k]   (W transposed)

  const int tid  = threadIdx.x;
  const int lane = tid & 31;
  const int wave = tid >> 5;
  const int wm   = wave >> 2;      // 0..1  (M direction, 64 rows each)
  const int wn   = wave & 3;       // 0..3  (N direction, 32 cols each)
  const int hi   = lane >> 4;      // half-wave select
  const int l16  = lane & 15;

  const int m0 = blockIdx.y * 128;
  const int n0 = blockIdx.x * 128;
  const size_t batch = blockIdx.z;

  const float*  Af = (const float*) Av + batch * (size_t)M * K;
  const __bf16* Ab = (const __bf16*)Av + batch * (size_t)M * K;

  v8f acc[4][2];
  #pragma unroll
  for (int mt = 0; mt < 4; ++mt)
    #pragma unroll
    for (int nt = 0; nt < 2; ++nt)
      #pragma unroll
      for (int i = 0; i < 8; ++i) acc[mt][nt][i] = 0.0f;

  const int arow = tid >> 1;        // 0..127
  const int akc  = (tid & 1) * 16;  // 0 or 16
  const int wkp  = tid >> 4;        // 0..15 -> W k-rows 2*wkp, 2*wkp+1
  const int wnc  = (tid & 15) * 8;  // 0..120 (n)

  for (int k0 = 0; k0 < K; k0 += 32) {
    __syncthreads();
    // ---- stage A tile (128x32) ----
    if (A_BF16) {
      // direct bf16 copy: async DMA to LDS (ASYNCcnt), 2 x 16B per thread
      const __bf16* gp = Ab + (size_t)(m0 + arow) * K + k0 + akc;
      const unsigned ldst = lds_off_of(&sA[arow][akc]);
      async_g2l_b128(ldst,      gp);
      async_g2l_b128(ldst + 16, gp + 8);
    } else {
      const float4* src = (const float4*)(Af + (size_t)(m0 + arow) * K + k0 + akc);
      #pragma unroll
      for (int v = 0; v < 4; ++v) {
        float4 f = src[v];
        sA[arow][akc + v*4 + 0] = f2bf(f.x);
        sA[arow][akc + v*4 + 1] = f2bf(f.y);
        sA[arow][akc + v*4 + 2] = f2bf(f.z);
        sA[arow][akc + v*4 + 3] = f2bf(f.w);
      }
    }
    // ---- stage W tile transposed (32x128 -> sB[n][k]), paired b32 stores ----
    {
      const float4* s0 = (const float4*)(W + (size_t)(k0 + 2*wkp)     * N + n0 + wnc);
      const float4* s1 = (const float4*)(W + (size_t)(k0 + 2*wkp + 1) * N + n0 + wnc);
      float4 a0 = s0[0], a1 = s0[1];
      float4 b0 = s1[0], b1 = s1[1];
      float e0[8] = {a0.x, a0.y, a0.z, a0.w, a1.x, a1.y, a1.z, a1.w};
      float e1[8] = {b0.x, b0.y, b0.z, b0.w, b1.x, b1.y, b1.z, b1.w};
      #pragma unroll
      for (int i = 0; i < 8; ++i) {
        const unsigned pk = bfbits(f2bf(e0[i])) | (bfbits(f2bf(e1[i])) << 16);
        *(unsigned*)&sB[wnc + i][2*wkp] = pk;   // two k's of one n-row at once
      }
    }
    // prefetch next A tile into caches
    if (k0 + 32 < K) {
      if (A_BF16) __builtin_prefetch(Ab + (size_t)(m0 + arow) * K + k0 + 32 + akc);
      else        __builtin_prefetch(Af + (size_t)(m0 + arow) * K + k0 + 32 + akc);
    }
    if (A_BF16) wait_async0();   // A DMA must land in LDS before the barrier
    __syncthreads();

    // ---- fragments + WMMA ----
    BFrag a[4], bfr[2];
    const int aklo = hi * 8;     // A layout: K in {klo..klo+7} U {klo+16..klo+23}
    #pragma unroll
    for (int mt = 0; mt < 4; ++mt) {
      const int r = wm*64 + mt*16 + l16;
      a[mt].h[0] = *(const v8bf*)&sA[r][aklo];
      a[mt].h[1] = *(const v8bf*)&sA[r][aklo + 16];
    }
    const int bklo = hi * 8;     // B layout: K in {klo..klo+7} U {klo+16..klo+23}
    #pragma unroll
    for (int nt = 0; nt < 2; ++nt) {
      const int c = wn*32 + nt*16 + l16;
      bfr[nt].h[0] = *(const v8bf*)&sB[c][bklo];
      bfr[nt].h[1] = *(const v8bf*)&sB[c][bklo + 16];
    }
    #pragma unroll
    for (int mt = 0; mt < 4; ++mt)
      #pragma unroll
      for (int nt = 0; nt < 2; ++nt)
        acc[mt][nt] = wmma_bf16(a[mt], bfr[nt], acc[mt][nt]);
  }

  // ---- epilogue: bias + store (C layout: VGPR i -> row +i+8*hi, lane -> col) ----
  #pragma unroll
  for (int mt = 0; mt < 4; ++mt)
    #pragma unroll
    for (int nt = 0; nt < 2; ++nt) {
      const int col = n0 + wn*32 + nt*16 + l16;
      const float bb = bias[col];
      #pragma unroll
      for (int i = 0; i < 8; ++i) {
        const int row = m0 + wm*64 + mt*16 + i + hi*8;
        const float vv = acc[mt][nt][i] + bb;
        const size_t idx = batch * (size_t)M * N + (size_t)row * N + col;
        if (OUT_F32) ((float*)Cv)[idx] = vv;
        else         ((__bf16*)Cv)[idx] = f2bf(vv);
      }
    }
}

// ---------------------------------------------------------------------------
// Flash attention: block = (batch, head, 64 q-rows), 4 waves x 16 rows.
// Q fragments straight from global; K chunk staged via async-to-LDS DMA
// (B-frag layout is contiguous in row-major K); V staged transposed into LDS
// with paired b32 stores; P through per-wave LDS (C->A relayout).
// ---------------------------------------------------------------------------
__global__ __launch_bounds__(128) void attn_kernel(
    const __bf16* __restrict__ Q, const __bf16* __restrict__ Kb,
    const __bf16* __restrict__ V, const float* __restrict__ mask,
    __bf16* __restrict__ O)
{
  __shared__ __bf16 sK [64][72];      // [ly-chunk][dk]     (async DMA target)
  __shared__ __bf16 sVt[64][72];      // [dk][ly-chunk]     (manual transpose)
  __shared__ __bf16 sP[4][16][72];    // per-wave P tile [qrow][ly-chunk]

  const int tid  = threadIdx.x;
  const int lane = tid & 31;
  const int wave = tid >> 5;
  const int hi   = lane >> 4;
  const int l16  = lane & 15;

  const int b = blockIdx.z, h = blockIdx.y;
  const int q0 = blockIdx.x * 64 + wave * 16;

  const __bf16* Qh = Q  + (size_t)b * LX * HD + h * DKK;
  const __bf16* Kh = Kb + (size_t)b * LY * HD + h * DKK;
  const __bf16* Vh = V  + (size_t)b * LY * HD + h * DKK;

  // Q fragments (A layout), kept in registers for the whole ly loop
  BFrag qa[2];
  #pragma unroll
  for (int j = 0; j < 2; ++j) {
    const __bf16* p = Qh + (size_t)(q0 + l16) * HD + j*32 + hi*8;
    qa[j].h[0] = *(const v8bf*)p;
    qa[j].h[1] = *(const v8bf*)(p + 16);
  }

  float mrow[8], lrow[8];
  v8f o[4];
  #pragma unroll
  for (int i = 0; i < 8; ++i) { mrow[i] = -3.0e38f; lrow[i] = 0.0f; }
  #pragma unroll
  for (int d = 0; d < 4; ++d)
    #pragma unroll
    for (int i = 0; i < 8; ++i) o[d][i] = 0.0f;

  const float scale = 0.125f;       // 1/sqrt(64)
  const int krow_st = tid >> 1;     // 0..63   (K staging row)
  const int kseg    = (tid & 1) * 32;
  const int vrp     = tid >> 2;     // 0..31 -> V rows 2*vrp, 2*vrp+1
  const int vdkb    = (tid & 3) * 16;

  for (int kb = 0; kb < LY; kb += 64) {
    __syncthreads();
    // ---- stage K chunk via async DMA: sK[kr][dk], 4 x 16B per thread ----
    {
      const __bf16* gp = Kh + (size_t)(kb + krow_st) * HD + kseg;
      const unsigned ldst = lds_off_of(&sK[krow_st][kseg]);
      #pragma unroll
      for (int g = 0; g < 4; ++g)
        async_g2l_b128(ldst + g*16, gp + g*8);
    }
    // ---- stage V chunk transposed: sVt[dk][kr], paired b32 stores ----
    {
      const __bf16* p0 = Vh + (size_t)(kb + 2*vrp) * HD + vdkb;
      union { uint4 u[2]; unsigned short s[16]; } r0, r1;
      r0.u[0] = *(const uint4*)(p0);       r0.u[1] = *(const uint4*)(p0 + 8);
      r1.u[0] = *(const uint4*)(p0 + HD);  r1.u[1] = *(const uint4*)(p0 + HD + 8);
      #pragma unroll
      for (int i = 0; i < 16; ++i) {
        const unsigned pk = (unsigned)r0.s[i] | ((unsigned)r1.s[i] << 16);
        *(unsigned*)&sVt[vdkb + i][2*vrp] = pk;
      }
    }
    wait_async0();       // K DMA in LDS before the barrier
    __syncthreads();

    // ---- scores: S = Q @ K^T (4 col-tiles of 16), + scale + mask ----
    float sreg[4][8];
    #pragma unroll
    for (int nt = 0; nt < 4; ++nt) {
      v8f s = {0,0,0,0,0,0,0,0};
      #pragma unroll
      for (int j = 0; j < 2; ++j) {
        BFrag kf;   // B layout: K(=dk) in {kk..kk+7} U {kk+16..kk+23}
        const int kk = j*32 + hi*8;
        kf.h[0] = *(const v8bf*)&sK[nt*16 + l16][kk];
        kf.h[1] = *(const v8bf*)&sK[nt*16 + l16][kk + 16];
        s = wmma_bf16(qa[j], kf, s);
      }
      const int kr = kb + nt*16 + l16;
      #pragma unroll
      for (int i = 0; i < 8; ++i) {
        const int qr = q0 + i + hi*8;
        sreg[nt][i] = s[i] * scale + mask[(size_t)b*LX*LY + (size_t)qr*LY + kr];
      }
    }

    // ---- online softmax (row stats align with C-layout: VGPR i <-> row) ----
    float mnew[8], sc[8], rs[8];
    #pragma unroll
    for (int i = 0; i < 8; ++i) {
      float v = fmaxf(fmaxf(sreg[0][i], sreg[1][i]), fmaxf(sreg[2][i], sreg[3][i]));
      v = fmaxf(v, __shfl_xor(v, 1, 32));
      v = fmaxf(v, __shfl_xor(v, 2, 32));
      v = fmaxf(v, __shfl_xor(v, 4, 32));
      v = fmaxf(v, __shfl_xor(v, 8, 32));
      mnew[i] = fmaxf(mrow[i], v);
      sc[i] = __expf(mrow[i] - mnew[i]);
      mrow[i] = mnew[i];
      rs[i] = 0.0f;
    }
    #pragma unroll
    for (int nt = 0; nt < 4; ++nt)
      #pragma unroll
      for (int i = 0; i < 8; ++i) {
        const float p = __expf(sreg[nt][i] - mnew[i]);
        rs[i] += p;
        sP[wave][i + hi*8][nt*16 + l16] = f2bf(p);   // C-layout scatter
      }
    #pragma unroll
    for (int i = 0; i < 8; ++i) {
      float r = rs[i];
      r += __shfl_xor(r, 1, 32);
      r += __shfl_xor(r, 2, 32);
      r += __shfl_xor(r, 4, 32);
      r += __shfl_xor(r, 8, 32);
      lrow[i] = lrow[i] * sc[i] + r;
      #pragma unroll
      for (int d = 0; d < 4; ++d) o[d][i] *= sc[i];
    }

    // ---- O += P @ V  (per-wave LDS read-back, same-wave DS order is safe) ----
    #pragma unroll
    for (int j = 0; j < 2; ++j) {
      BFrag pa;
      const int pk = j*32 + hi*8;
      pa.h[0] = *(const v8bf*)&sP[wave][l16][pk];
      pa.h[1] = *(const v8bf*)&sP[wave][l16][pk + 16];
      #pragma unroll
      for (int d = 0; d < 4; ++d) {
        BFrag vb;
        const int vk = j*32 + hi*8;
        vb.h[0] = *(const v8bf*)&sVt[d*16 + l16][vk];
        vb.h[1] = *(const v8bf*)&sVt[d*16 + l16][vk + 16];
        o[d] = wmma_bf16(pa, vb, o[d]);
      }
    }
  }

  // ---- normalize + store attn output (bf16) into workspace ----
  #pragma unroll
  for (int d = 0; d < 4; ++d)
    #pragma unroll
    for (int i = 0; i < 8; ++i) {
      const int qr = q0 + i + hi*8;
      const float val = o[d][i] / lrow[i];
      O[((size_t)b*LX + qr) * HD + h*DKK + d*16 + l16] = f2bf(val);
    }
}

// ---------------------------------------------------------------------------
extern "C" void kernel_launch(void* const* d_in, const int* in_sizes, int n_in,
                              void* d_out, int out_size, void* d_ws, size_t ws_size,
                              hipStream_t stream) {
  (void)in_sizes; (void)n_in; (void)out_size; (void)ws_size;
  const float* x    = (const float*)d_in[0];
  const float* y    = (const float*)d_in[1];
  const float* mask = (const float*)d_in[2];
  const float* Wq   = (const float*)d_in[3];
  const float* bq   = (const float*)d_in[4];
  const float* Wk   = (const float*)d_in[5];
  const float* bk   = (const float*)d_in[6];
  const float* Wv   = (const float*)d_in[7];
  const float* bv   = (const float*)d_in[8];
  const float* Wo   = (const float*)d_in[9];
  const float* bo   = (const float*)d_in[10];

  const size_t per = (size_t)BS * LX * HD;     // elements per bf16 buffer
  __bf16* qws = (__bf16*)d_ws;
  __bf16* kws = qws + per;
  __bf16* vws = kws + per;
  __bf16* aws = vws + per;

  dim3 gblk(256);
  dim3 gg(HD/128, LX/128, BS);
  gemm_kernel<false,false><<<gg, gblk, 0, stream>>>(x, Wq, bq, qws, LX, HD, DD);
  gemm_kernel<false,false><<<gg, gblk, 0, stream>>>(y, Wk, bk, kws, LY, HD, DD);
  gemm_kernel<false,false><<<gg, gblk, 0, stream>>>(y, Wv, bv, vws, LY, HD, DD);

  dim3 ag(LX/64, HH, BS), ab(128);
  attn_kernel<<<ag, ab, 0, stream>>>(qws, kws, vws, mask, aws);

  dim3 og(DD/128, LX/128, BS);
  gemm_kernel<true,true><<<og, gblk, 0, stream>>>((const void*)aws, Wo, bo, d_out, LX, DD, HD);
}
